// SparseCoder_32315333935755
// MI455X (gfx1250) — matrix operand order repaired and checked by reading, not compiled, not run on hardware
//
#include <hip/hip_runtime.h>

// ---------------- problem constants ----------------
#define NROWS 8192
#define DIN   1024
#define NLAT  32768
#define TOPK  64

#define ROWS_PER_WG 32          // rows handled by one workgroup
#define CHUNK 256               // latent columns processed per tile
#define NCHUNK (NLAT / CHUNK)   // 128
#define LDSTRIDE 260            // floats; avoids DS bank conflicts (8*260 % 64 == 32)

// output layout (float32, concatenated in reference return order)
#define ACT_OFF ((size_t)NROWS * DIN)                 // 8388608
#define IDX_OFF (ACT_OFF + (size_t)NROWS * TOPK)      // 8912896
#define FVU_OFF (IDX_OFF + (size_t)NROWS * TOPK)      // 9437184

// workspace layout (bytes)
#define XB_BYTES ((size_t)NROWS * DIN * 2)            // 16 MB bf16 (x - b_dec)
#define WB_BYTES ((size_t)NLAT  * DIN * 2)            // 64 MB bf16 W_enc
#define CM_OFF   (XB_BYTES + WB_BYTES)
#define ACC_OFF  (CM_OFF + 4096)

typedef __bf16 v4bf  __attribute__((ext_vector_type(4)));
typedef __bf16 v8bf  __attribute__((ext_vector_type(8)));
typedef __bf16 v16bf __attribute__((ext_vector_type(16)));
typedef float  v8f   __attribute__((ext_vector_type(8)));

#define NEG_INF (-__builtin_inff())

// f32 -> bf16 round-to-nearest-even (inputs are finite)
__device__ inline unsigned short f2bf(float f) {
    unsigned u = __builtin_bit_cast(unsigned, f);
    u += 0x7fffu + ((u >> 16) & 1u);
    return (unsigned short)(u >> 16);
}

// ---------------- kernel 0: convert inputs to bf16 workspace ----------------
__global__ __launch_bounds__(256) void k_convert(const float* __restrict__ x,
                                                 const float* __restrict__ W_enc,
                                                 const float* __restrict__ b_dec,
                                                 unsigned short* __restrict__ xb,
                                                 unsigned short* __restrict__ wb,
                                                 float* __restrict__ accum) {
    if (blockIdx.x == 0 && threadIdx.x == 0) { accum[0] = 0.f; accum[1] = 0.f; }
    const int stride = gridDim.x * blockDim.x;
    const int gid = blockIdx.x * blockDim.x + threadIdx.x;
    const int totW4 = (NLAT * DIN) / 4;     // fits in int
    const int totX4 = (NROWS * DIN) / 4;
    for (int i = gid; i < totW4; i += stride) {
        float4 w = ((const float4*)W_enc)[i];
        ushort4 o; o.x = f2bf(w.x); o.y = f2bf(w.y); o.z = f2bf(w.z); o.w = f2bf(w.w);
        ((ushort4*)wb)[i] = o;
    }
    for (int i = gid; i < totX4; i += stride) {
        float4 v = ((const float4*)x)[i];
        const int bc = i & (DIN / 4 - 1);
        float4 b = ((const float4*)b_dec)[bc];
        ushort4 o;
        o.x = f2bf(v.x - b.x); o.y = f2bf(v.y - b.y);
        o.z = f2bf(v.z - b.z); o.w = f2bf(v.w - b.w);
        ((ushort4*)xb)[i] = o;
    }
}

// ---------------- kernel 1: fused bf16-WMMA encoder GEMM + per-row top-64 ----------------
__global__ __launch_bounds__(512) void k_encode_topk(const unsigned short* __restrict__ xb_,
                                                     const unsigned short* __restrict__ wb_,
                                                     const float* __restrict__ b_enc,
                                                     float* __restrict__ outAct,
                                                     float* __restrict__ outIdx) {
    __shared__ float lds_pre[ROWS_PER_WG * LDSTRIDE];

    const __bf16* xb = (const __bf16*)xb_;
    const __bf16* wb = (const __bf16*)wb_;

    const int tid  = threadIdx.x;
    const int lane = tid & 31;
    const int wv   = tid >> 5;           // 16 waves
    const int half = lane >> 4;          // 0: lanes 0-15, 1: lanes 16-31
    const int ln   = lane & 15;
    const int rowBase = blockIdx.x * ROWS_PER_WG;

    // A fragment base pointers for the two M tiles (contiguous k order, same as B):
    // lane holds K in {16*half .. 16*half+15}
    const __bf16* A0 = xb + (size_t)(rowBase + ln) * DIN + half * 16;
    const __bf16* A1 = A0 + (size_t)16 * DIN;

    // running top-64 state: wave wv owns local rows wv and wv+16; 2 slots per lane
    float tv0[2] = {-1.f, -1.f}, tv1[2] = {-1.f, -1.f};
    int   ti0[2] = {0, 0},       ti1[2] = {0, 0};
    float mn[2]  = {-1.f, -1.f};
    int   mlane[2] = {0, 0}, mslot[2] = {0, 0};

    for (int ch = 0; ch < NCHUNK; ++ch) {
        const int colTile = ch * CHUNK;
        const int col = colTile + wv * 16 + ln;
        // B fragment: lane n holds 16 consecutive K values of W_enc row `col`
        const __bf16* B = wb + (size_t)col * DIN + half * 16;
        if (ch + 1 < NCHUNK)  // pull next chunk's W_enc rows toward L2/L0
            __builtin_prefetch(wb + (size_t)(col + CHUNK) * DIN + half * 16, 0, 1);

        v8f c0 = {}; v8f c1 = {};
        #pragma unroll 4
        for (int kc = 0; kc < DIN; kc += 32) {
            v8bf a0lo = *(const v8bf*)(const void*)(A0 + kc);
            v8bf a0hi = *(const v8bf*)(const void*)(A0 + kc + 8);
            v16bf a0 = __builtin_shufflevector(a0lo, a0hi, 0,1,2,3,4,5,6,7,8,9,10,11,12,13,14,15);
            v16bf bf = *(const v16bf*)(const void*)(B + kc);
            c0 = __builtin_amdgcn_wmma_f32_16x16x32_bf16(false, a0, false, bf, (short)0, c0, false, false);
            v8bf a1lo = *(const v8bf*)(const void*)(A1 + kc);
            v8bf a1hi = *(const v8bf*)(const void*)(A1 + kc + 8);
            v16bf a1 = __builtin_shufflevector(a1lo, a1hi, 0,1,2,3,4,5,6,7,8,9,10,11,12,13,14,15);
            c1 = __builtin_amdgcn_wmma_f32_16x16x32_bf16(false, a1, false, bf, (short)0, c1, false, false);
        }

        // bias + ReLU, stage 32x256 tile in LDS. C layout: VGPR r -> M = r + 8*half.
        const float be = b_enc[col];
        #pragma unroll
        for (int r = 0; r < 8; ++r) {
            const int m = r + half * 8;
            float v0 = c0[r] + be; v0 = v0 > 0.f ? v0 : 0.f;
            float v1 = c1[r] + be; v1 = v1 > 0.f ? v1 : 0.f;
            lds_pre[m * LDSTRIDE + wv * 16 + ln]        = v0;
            lds_pre[(m + 16) * LDSTRIDE + wv * 16 + ln] = v1;
        }
        __syncthreads();

        // phase 2: merge chunk into running top-64 (wave wv -> rows wv, wv+16)
        #pragma unroll
        for (int rr = 0; rr < 2; ++rr) {
            const int lrow = wv + rr * 16;
            #pragma unroll
            for (int it = 0; it < 8; ++it) {
                float v  = lds_pre[lrow * LDSTRIDE + it * 32 + lane];
                int   ci = colTile + it * 32 + lane;
                while (true) {
                    unsigned m = (unsigned)__ballot(v > mn[rr]);
                    if (!m) break;
                    const int src = __builtin_ctz(m);
                    const float cv = __shfl(v, src, 32);
                    const int   cc = __shfl(ci, src, 32);
                    if (lane == src) v = NEG_INF;          // candidate consumed
                    if (lane == mlane[rr]) {               // replace current min
                        if (mslot[rr]) { tv1[rr] = cv; ti1[rr] = cc; }
                        else           { tv0[rr] = cv; ti0[rr] = cc; }
                    }
                    // argmin butterfly over the 64 slots
                    const bool s = tv1[rr] < tv0[rr];
                    float mv = s ? tv1[rr] : tv0[rr];
                    int meta = (lane << 1) | (s ? 1 : 0);
                    #pragma unroll
                    for (int off = 16; off >= 1; off >>= 1) {
                        const float ov = __shfl_xor(mv, off, 32);
                        const int   om = __shfl_xor(meta, off, 32);
                        if (ov < mv || (ov == mv && om < meta)) { mv = ov; meta = om; }
                    }
                    mn[rr] = mv; mlane[rr] = meta >> 1; mslot[rr] = meta & 1;
                }
            }
        }
        __syncthreads();
    }

    // selection sort (descending, ties -> lower index, matching lax.top_k) + write
    #pragma unroll
    for (int rr = 0; rr < 2; ++rr) {
        const int grow = rowBase + wv + rr * 16;
        float* oa = outAct + (size_t)grow * TOPK;
        float* oi = outIdx + (size_t)grow * TOPK;
        for (int k = 0; k < TOPK; ++k) {
            const bool s = (tv1[rr] > tv0[rr]) || (tv1[rr] == tv0[rr] && ti1[rr] < ti0[rr]);
            float mv = s ? tv1[rr] : tv0[rr];
            int   mc = s ? ti1[rr] : ti0[rr];
            int meta = (lane << 1) | (s ? 1 : 0);
            #pragma unroll
            for (int off = 16; off >= 1; off >>= 1) {
                const float ov = __shfl_xor(mv, off, 32);
                const int   oc = __shfl_xor(mc, off, 32);
                const int   om = __shfl_xor(meta, off, 32);
                if (ov > mv || (ov == mv && oc < mc)) { mv = ov; mc = oc; meta = om; }
            }
            if (lane == 0) { oa[k] = mv; oi[k] = (float)mc; }
            if (lane == (meta >> 1)) {
                if (meta & 1) tv1[rr] = NEG_INF; else tv0[rr] = NEG_INF;
            }
        }
    }
}

// ---------------- kernel 2: sparse decode (gather 64 decoder rows per token) ----------------
__global__ __launch_bounds__(256) void k_decode(const float* __restrict__ acts,
                                                const float* __restrict__ idxf,
                                                const float* __restrict__ W_dec,
                                                const float* __restrict__ b_dec,
                                                float* __restrict__ out) {
    __shared__ float sA[TOPK];
    __shared__ int   sI[TOPK];
    const int n = blockIdx.x;
    const int t = threadIdx.x;
    if (t < TOPK) {
        sA[t] = acts[(size_t)n * TOPK + t];
        sI[t] = (int)idxf[(size_t)n * TOPK + t];
    }
    __syncthreads();
    const int c = t * 4;
    float4 acc = *(const float4*)(b_dec + c);
    #pragma unroll 4
    for (int k = 0; k < TOPK; ++k) {
        const float a = sA[k];
        const float4 w = *(const float4*)(W_dec + (size_t)sI[k] * DIN + c);
        acc.x += a * w.x; acc.y += a * w.y; acc.z += a * w.z; acc.w += a * w.w;
    }
    *(float4*)(out + (size_t)n * DIN + c) = acc;
}

// ---------------- kernel 3a: per-column mean of x ----------------
__global__ __launch_bounds__(256) void k_colmean(const float* __restrict__ x,
                                                 float* __restrict__ colmean) {
    __shared__ float red[256];
    const int d = blockIdx.x;
    const int t = threadIdx.x;
    float s = 0.f;
    for (int r = t; r < NROWS; r += 256) s += x[(size_t)r * DIN + d];
    red[t] = s; __syncthreads();
    for (int o = 128; o > 0; o >>= 1) { if (t < o) red[t] += red[t + o]; __syncthreads(); }
    if (t == 0) colmean[d] = red[0] / (float)NROWS;
}

// ---------------- kernel 3b: variance + squared-error sums ----------------
__global__ __launch_bounds__(256) void k_sums(const float* __restrict__ x,
                                              const float* __restrict__ sae,
                                              const float* __restrict__ colmean,
                                              float* __restrict__ accum) {
    __shared__ float rv[256], re[256];
    const size_t total = (size_t)NROWS * DIN;
    float sv = 0.f, se = 0.f;
    for (size_t i = (size_t)blockIdx.x * 256 + threadIdx.x; i < total;
         i += (size_t)gridDim.x * 256) {
        const float xv = x[i];
        const float dv = xv - colmean[i & (DIN - 1)];
        const float ev = xv - sae[i];
        sv += dv * dv; se += ev * ev;
    }
    const int t = threadIdx.x;
    rv[t] = sv; re[t] = se; __syncthreads();
    for (int o = 128; o > 0; o >>= 1) {
        if (t < o) { rv[t] += rv[t + o]; re[t] += re[t + o]; }
        __syncthreads();
    }
    if (t == 0) { atomicAdd(&accum[0], rv[0]); atomicAdd(&accum[1], re[0]); }
}

// ---------------- kernel 3c: fvu + auxk ----------------
__global__ void k_final(const float* __restrict__ accum, float* __restrict__ out_tail) {
    if (threadIdx.x == 0) {
        out_tail[0] = accum[1] / accum[0];   // fvu
        out_tail[1] = 0.f;                   // auxk_loss
    }
}

// ---------------- launch ----------------
extern "C" void kernel_launch(void* const* d_in, const int* in_sizes, int n_in,
                              void* d_out, int out_size, void* d_ws, size_t ws_size,
                              hipStream_t stream) {
    const float* x     = (const float*)d_in[0];
    const float* W_enc = (const float*)d_in[1];
    const float* b_enc = (const float*)d_in[2];
    const float* W_dec = (const float*)d_in[3];
    const float* b_dec = (const float*)d_in[4];
    float* out = (float*)d_out;

    char* ws = (char*)d_ws;
    unsigned short* xb = (unsigned short*)ws;
    unsigned short* wb = (unsigned short*)(ws + XB_BYTES);
    float* colmean = (float*)(ws + CM_OFF);
    float* accum   = (float*)(ws + ACC_OFF);

    float* outAct = out + ACT_OFF;
    float* outIdx = out + IDX_OFF;
    float* outTail = out + FVU_OFF;

    k_convert<<<2048, 256, 0, stream>>>(x, W_enc, b_dec, xb, wb, accum);
    k_encode_topk<<<NROWS / ROWS_PER_WG, 512, 0, stream>>>(xb, wb, b_enc, outAct, outIdx);
    k_decode<<<NROWS, 256, 0, stream>>>(outAct, outIdx, W_dec, b_dec, out);
    k_colmean<<<DIN, 256, 0, stream>>>(x, colmean);
    k_sums<<<2048, 256, 0, stream>>>(x, out, colmean, accum);
    k_final<<<1, 32, 0, stream>>>(accum, outTail);
}
